// ClassLayer_56564719289025
// MI455X (gfx1250) — compile-verified, matrix operand order checked by reading, not run on hardware
//
#include <hip/hip_runtime.h>
#include <hip/hip_bf16.h>

typedef __attribute__((ext_vector_type(16))) _Float16 v16h;
typedef __attribute__((ext_vector_type(8)))  _Float16 v8h;
typedef __attribute__((ext_vector_type(4)))  _Float16 v4h;
typedef __attribute__((ext_vector_type(8)))  float    v8f;
typedef __attribute__((ext_vector_type(4)))  float    v4f;

#define M_DIM 8192
#define N_DIM 4096
#define K_DIM 4096

#define BM 128
#define BN 128
#define BK 64
#define LDSS 72           // padded row stride in halfs (144B): conflict-free b128 reads
#define TILE (BM * LDSS)  // halfs per staged tile (A or B)

typedef __attribute__((address_space(3))) _Float16 lds_f16;

// ---------------- CDNA5 async global->LDS copy (ASYNCcnt path) ----------------
// Pass the REAL LDS pointer (addrspace(3), 32-bit) so the shared array escapes
// into the asm: LLVM must then honor the "memory" clobber for LDS and cannot
// fold the subsequent ds_load fragment reads to undef.
__device__ __forceinline__ void async_copy_b128(_Float16* lds_ptr, const void* gptr) {
    lds_f16* p3 = (lds_f16*)lds_ptr;   // addrspacecast generic -> LDS (32-bit)
    asm volatile("global_load_async_to_lds_b128 %0, %1, off"
                 :: "v"(p3), "v"(gptr)
                 : "memory");
}

__device__ __forceinline__ void wait_async0() {
    asm volatile("s_wait_asynccnt 0" ::: "memory");
}

// ---------------- scale = mean(|W|), deterministic two-stage reduction ----------------
__global__ __launch_bounds__(256) void reduce_abs_partial(const float* __restrict__ w,
                                                          float* __restrict__ partials,
                                                          int n) {
    __shared__ float sm[256];
    float s = 0.0f;
    for (int i = blockIdx.x * blockDim.x + threadIdx.x; i < n; i += gridDim.x * blockDim.x)
        s += fabsf(w[i]);
    sm[threadIdx.x] = s;
    __syncthreads();
    for (int off = 128; off > 0; off >>= 1) {
        if (threadIdx.x < (unsigned)off) sm[threadIdx.x] += sm[threadIdx.x + off];
        __syncthreads();
    }
    if (threadIdx.x == 0) partials[blockIdx.x] = sm[0];
}

__global__ __launch_bounds__(256) void reduce_final(const float* __restrict__ partials,
                                                    float* __restrict__ scale,
                                                    int np, float inv_count) {
    __shared__ float sm[256];
    float s = 0.0f;
    for (int i = threadIdx.x; i < np; i += 256) s += partials[i];
    sm[threadIdx.x] = s;
    __syncthreads();
    for (int off = 128; off > 0; off >>= 1) {
        if (threadIdx.x < (unsigned)off) sm[threadIdx.x] += sm[threadIdx.x + off];
        __syncthreads();
    }
    if (threadIdx.x == 0) *scale = sm[0] * inv_count;
}

// ---------------- pack x (f32 -> f16), vectorized ----------------
__global__ __launch_bounds__(256) void pack_x(const float* __restrict__ x,
                                              _Float16* __restrict__ xh, int n4) {
    for (int i = blockIdx.x * blockDim.x + threadIdx.x; i < n4; i += gridDim.x * blockDim.x) {
        v4f v = *(const v4f*)(x + (size_t)i * 4);
        v4h h;
        h.x = (_Float16)v.x; h.y = (_Float16)v.y;
        h.z = (_Float16)v.z; h.w = (_Float16)v.w;
        *(v4h*)(xh + (size_t)i * 4) = h;
    }
}

// ---------------- pack sign(W) (f32 -> f16 in {-1,0,+1}) ----------------
__global__ __launch_bounds__(256) void pack_w_sign(const float* __restrict__ w,
                                                   _Float16* __restrict__ wh, int n4) {
    for (int i = blockIdx.x * blockDim.x + threadIdx.x; i < n4; i += gridDim.x * blockDim.x) {
        v4f v = *(const v4f*)(w + (size_t)i * 4);
        v4h h;
        #pragma unroll
        for (int c = 0; c < 4; ++c) {
            float f = v[c];
            h[c] = (_Float16)((f > 0.0f) ? 1.0f : ((f < 0.0f) ? -1.0f : 0.0f));
        }
        *(v4h*)(wh + (size_t)i * 4) = h;
    }
}

// ---------------- WMMA GEMM: Y[m,n] = scale * sum_k Xh[m,k] * Wh[n,k] ----------------
// Block tile 128x128, K-step 64, double-buffered LDS staged via async-to-LDS,
// pipeline unrolled x2 so buffer indices are compile-time constants.
// 8 waves (wave32) in 2x4 grid; each wave computes 4(M) x 2(N) 16x16 tiles,
// 16 v_wmma_f32_16x16x32_f16 per K-tile with f32 accumulators.
__global__ __launch_bounds__(256) void gemm_bin_f16(const _Float16* __restrict__ Xh,
                                                    const _Float16* __restrict__ Wh,
                                                    const float* __restrict__ scale_p,
                                                    float* __restrict__ Y) {
    // layout: [buf0 A][buf0 B][buf1 A][buf1 B], each TILE halfs -> 72 KB total
    __shared__ _Float16 SMEM[4 * TILE];

    // force SMEM to escape so asm memory clobbers must be assumed to write it
    asm volatile("" :: "v"(&SMEM[0]) : "memory");

    const int tid  = threadIdx.x;
    const int lane = tid & 31;
    const int wave = tid >> 5;     // 0..7
    const int wm   = wave >> 2;    // 0..1  -> 64-row slab
    const int wn   = wave & 3;     // 0..3  -> 32-col slab

    const int block_m = blockIdx.y * BM;
    const int block_n = blockIdx.x * BN;

    // fragment addressing (ISA 7.12.2, 16-bit A 16x32 layout; B symmetric for row-major [N,K])
    const int fr  = lane & 15;               // row (A) / col (B) within 16x16 tile
    const int kho = (lane < 16) ? 0 : 8;     // K half-offset for this lane

    // staging assignment: thread t covers rows (t>>3)+32r, col chunk (t&7)*8 halfs
    const int srow = tid >> 3;               // 0..31
    const int scol = (tid & 7) * 8;          // 0..56

    // precomputed per-thread staging bases
    const _Float16* gA = Xh + (size_t)(block_m + srow) * K_DIM + scol;
    const _Float16* gB = Wh + (size_t)(block_n + srow) * K_DIM + scol;
    const size_t grstride = (size_t)32 * K_DIM;          // 32-row global stride (halfs)
    _Float16* lbase = SMEM + srow * LDSS + scol;         // LDS staging base (this thread)

    v8f acc[4][2];
    #pragma unroll
    for (int i = 0; i < 4; ++i)
        #pragma unroll
        for (int j = 0; j < 2; ++j)
            acc[i][j] = (v8f){0.f, 0.f, 0.f, 0.f, 0.f, 0.f, 0.f, 0.f};

    // issue async stage of one K-tile into buffer BUFC (compile-time constant)
    auto stage = [&](const int BUFC, int k0) __attribute__((always_inline)) {
        _Float16* la = lbase + BUFC * 2 * TILE;
        _Float16* lb = la + TILE;
        #pragma unroll
        for (int r = 0; r < 4; ++r) {
            async_copy_b128(la + r * 32 * LDSS, gA + k0 + r * grstride);
            async_copy_b128(lb + r * 32 * LDSS, gB + k0 + r * grstride);
        }
    };

    // 16 WMMAs on buffer BUFC (compile-time constant)
    auto compute = [&](const int BUFC) __attribute__((always_inline)) {
        const _Float16* Abuf = SMEM + BUFC * 2 * TILE;
        const _Float16* Bbuf = Abuf + TILE;
        #pragma unroll
        for (int kk = 0; kk < BK; kk += 32) {
            v16h bfrag[2];
            #pragma unroll
            for (int j = 0; j < 2; ++j) {
                const _Float16* p = Bbuf + (wn * 32 + j * 16 + fr) * LDSS + kk + kho;
                v8h lo = *(const v8h*)(p);
                v8h hi = *(const v8h*)(p + 16);
                bfrag[j] = __builtin_shufflevector(lo, hi,
                    0, 1, 2, 3, 4, 5, 6, 7, 8, 9, 10, 11, 12, 13, 14, 15);
            }
            #pragma unroll
            for (int i = 0; i < 4; ++i) {
                const _Float16* p = Abuf + (wm * 64 + i * 16 + fr) * LDSS + kk + kho;
                v8h lo = *(const v8h*)(p);
                v8h hi = *(const v8h*)(p + 16);
                v16h afrag = __builtin_shufflevector(lo, hi,
                    0, 1, 2, 3, 4, 5, 6, 7, 8, 9, 10, 11, 12, 13, 14, 15);
                #pragma unroll
                for (int j = 0; j < 2; ++j) {
                    acc[i][j] = __builtin_amdgcn_wmma_f32_16x16x32_f16(
                        false, afrag, false, bfrag[j], (short)0, acc[i][j], false, false);
                }
            }
        }
    };

    stage(0, 0);  // prologue

    const int NITER = K_DIM / BK;  // 64 (even)
    for (int it = 0; it < NITER; it += 2) {
        // ---- half A: consume buf0, prefetch buf1 ----
        wait_async0();     // my buf0 writes landed
        __syncthreads();   // all waves: buf0 complete; all done reading buf1
        stage(1, (it + 1) * BK);            // it+1 <= NITER-1 always
        compute(0);

        // ---- half B: consume buf1, prefetch buf0 ----
        wait_async0();
        __syncthreads();
        if (it + 2 < NITER) stage(0, (it + 2) * BK);
        compute(1);
    }

    const float s = *scale_p;

    // epilogue: C/D layout — VGPR r: lanes 0-15 -> row r, lanes 16-31 -> row r+8; col = lane&15
    #pragma unroll
    for (int i = 0; i < 4; ++i) {
        #pragma unroll
        for (int j = 0; j < 2; ++j) {
            const int mbase = block_m + wm * 64 + i * 16 + ((lane >= 16) ? 8 : 0);
            const int nb    = block_n + wn * 32 + j * 16 + fr;
            #pragma unroll
            for (int r = 0; r < 8; ++r) {
                Y[(size_t)(mbase + r) * N_DIM + nb] = acc[i][j][r] * s;
            }
        }
    }
}

// ---------------- launch ----------------
extern "C" void kernel_launch(void* const* d_in, const int* in_sizes, int n_in,
                              void* d_out, int out_size, void* d_ws, size_t ws_size,
                              hipStream_t stream) {
    const float* x = (const float*)d_in[0];
    const float* w = (const float*)d_in[1];
    float* out = (float*)d_out;

    char* ws = (char*)d_ws;
    _Float16* xh = (_Float16*)ws;                                   // 64 MB
    _Float16* wh = (_Float16*)(ws + (size_t)M_DIM * K_DIM * 2);     // 32 MB
    float* partials = (float*)(ws + (size_t)M_DIM * K_DIM * 2
                                  + (size_t)N_DIM * K_DIM * 2);     // 4 KB
    float* scale = partials + 1024;

    const int nW = N_DIM * K_DIM;       // 16,777,216
    const int nX = M_DIM * K_DIM;       // 33,554,432

    reduce_abs_partial<<<1024, 256, 0, stream>>>(w, partials, nW);
    reduce_final<<<1, 256, 0, stream>>>(partials, scale, 1024, 1.0f / (float)nW);

    pack_x<<<2048, 256, 0, stream>>>(x, xh, nX / 4);
    pack_w_sign<<<2048, 256, 0, stream>>>(w, wh, nW / 4);

    dim3 grid(N_DIM / BN, M_DIM / BM);   // 32 x 64
    gemm_bin_f16<<<grid, 256, 0, stream>>>(xh, wh, scale, out);
}
